// Scorer_22351009809137
// MI455X (gfx1250) — matrix operand order repaired and checked by reading, not compiled, not run on hardware
//
#include <hip/hip_runtime.h>
#include <hip/hip_bf16.h>

// Problem constants
#define NN    1024     // sentence rows
#define HH    1024     // hidden dim
#define MLPH  100      // MLP hidden
#define NPAD  112      // MLPH padded to 7*16 for WMMA tiles
#define JTOT  1025     // 1 + N columns of raw/score matrix

typedef _Float16 v16h __attribute__((ext_vector_type(16)));
typedef float    v8f  __attribute__((ext_vector_type(8)));

// Fast tanh on the TRANS pipe (v_tanh_f32).
__device__ inline float tanh_fast(float x) {
#if __has_builtin(__builtin_amdgcn_tanhf)
    return __builtin_amdgcn_tanhf(x);
#else
    float r;
    asm("v_tanh_f32 %0, %1\n\tv_nop" : "=v"(r) : "v"(x));
    return r;
#endif
}

// Convert 8+8 contiguous f32 (at p0 and p1) into a packed v16h.
__device__ inline v16h cvt_2x8_f16(const float* __restrict__ p0,
                                   const float* __restrict__ p1) {
    float4 a0 = *(const float4*)(p0);
    float4 a1 = *(const float4*)(p0 + 4);
    float4 a2 = *(const float4*)(p1);
    float4 a3 = *(const float4*)(p1 + 4);
    v16h r;
    r[0]=(_Float16)a0.x;  r[1]=(_Float16)a0.y;  r[2]=(_Float16)a0.z;  r[3]=(_Float16)a0.w;
    r[4]=(_Float16)a1.x;  r[5]=(_Float16)a1.y;  r[6]=(_Float16)a1.z;  r[7]=(_Float16)a1.w;
    r[8]=(_Float16)a2.x;  r[9]=(_Float16)a2.y;  r[10]=(_Float16)a2.z; r[11]=(_Float16)a2.w;
    r[12]=(_Float16)a3.x; r[13]=(_Float16)a3.y; r[14]=(_Float16)a3.z; r[15]=(_Float16)a3.w;
    return r;
}

// ---------------------------------------------------------------------------
// Kernel A: XA' = 0.5*(sentence @ Wa.T + b1),  XB' = 0.5*(sentence @ Wb.T)
// One wave per 16x16 output tile; f16 WMMA, f32 accumulate; dual accumulators.
// grid = (64 m-tiles, 7 n-tiles, 2 matrices), block = 32.
// ---------------------------------------------------------------------------
__global__ __launch_bounds__(32)
void gemm_xab_wmma(const float* __restrict__ S,    // [1024][1024]
                   const float* __restrict__ W1,   // [100][2048]
                   const float* __restrict__ bias1,// [100]
                   float* __restrict__ XA,         // [1024][112]
                   float* __restrict__ XB)         // [1024][112]
{
    const int lane = threadIdx.x;        // 0..31
    const int lo   = lane & 15;
    const int hi   = lane >> 4;          // lane half
    const int mt   = blockIdx.x;         // 0..63
    const int nt   = blockIdx.y;         // 0..6
    const int mat  = blockIdx.z;         // 0 -> Wa/XA, 1 -> Wb/XB

    const int  n      = nt * 16 + lo;               // output column / W1 row
    const bool nvalid = (n < MLPH);
    const int  nc     = nvalid ? n : (MLPH - 1);    // clamp: garbage cols masked at store

    const float* srow = S  + (size_t)(mt * 16 + lo) * HH;
    const float* wrow = W1 + (size_t)nc * (2 * HH) + (size_t)mat * HH;

    v8f acc0 = {}, acc1 = {};

    for (int k0 = 0; k0 < HH; k0 += 64) {
        // A tile (16x32): half 0 -> K{0..7,16..23}, half 1 -> K{8..15,24..31}
        // B tile (32x16): column N = lo, half 0 -> K{0..7,16..23}, half 1 -> K{8..15,24..31}
        {
            const float* ap = srow + k0 + hi * 8;
            const float* bp = wrow + k0 + hi * 8;
            v16h A = cvt_2x8_f16(ap, ap + 16);
            v16h B = cvt_2x8_f16(bp, bp + 16);
            acc0 = __builtin_amdgcn_wmma_f32_16x16x32_f16(
                       false, A, false, B, (short)0, acc0, false, false);
        }
        {
            const float* ap = srow + k0 + 32 + hi * 8;
            const float* bp = wrow + k0 + 32 + hi * 8;
            v16h A = cvt_2x8_f16(ap, ap + 16);
            v16h B = cvt_2x8_f16(bp, bp + 16);
            acc1 = __builtin_amdgcn_wmma_f32_16x16x32_f16(
                       false, A, false, B, (short)0, acc1, false, false);
        }
    }
    v8f acc = acc0 + acc1;

    // D layout: VGPR v -> M = v + 8*hi, N = lo.  Fold b1 and the 0.5 scale.
    float* out = mat ? XB : XA;
    const float badd = (mat == 0 && nvalid) ? bias1[n] : 0.f;
    #pragma unroll
    for (int v = 0; v < 8; ++v) {
        const int row = mt * 16 + v + hi * 8;
        const float val = nvalid ? (acc[v] + badd) * 0.5f : 0.f;
        out[(size_t)row * NPAD + n] = val;
    }
}

// ---------------------------------------------------------------------------
// Kernel B: raw[i,j] = base + sum_k w2h[k] * tanh(XA'[i,k] + xb'_j[k])
// where w2h = 0.5*W2, base = b2 + sum(w2h).  One v_tanh_f32 per element.
// grid = (64 i-tiles, 5 j-chunks of 256), block = 256.
// ---------------------------------------------------------------------------
__global__ __launch_bounds__(256)
void raw_pass(const float* __restrict__ XA,
              const float* __restrict__ XB,
              const float* __restrict__ W2,
              const float* __restrict__ b2,
              float* __restrict__ raw_out)   // [1024][1025]
{
    __shared__ __align__(16) float xa_s[16 * MLPH];  // stride 100: 16 rows, distinct banks
    __shared__ __align__(16) float xb_s[16 * MLPH];
    __shared__ __align__(16) float w2h_s[MLPH];

    const int t     = threadIdx.x;
    const int iloc  = t & 15;
    const int jslot = t >> 4;           // 0..15
    const int ibase = blockIdx.x * 16;

    for (int idx = t; idx < 16 * MLPH; idx += 256) {
        const int r = idx / MLPH, c = idx - r * MLPH;
        xa_s[idx] = XA[(size_t)(ibase + r) * NPAD + c];
    }
    if (t < MLPH) w2h_s[t] = 0.5f * W2[t];
    const float b2v = b2[0];
    __syncthreads();

    float w2sum = 0.f;
    for (int k = 0; k < MLPH; ++k) w2sum += w2h_s[k];
    const float base = b2v + w2sum;

    const float* xarow = xa_s + iloc * MLPH;

    for (int pass = 0; pass < 16; ++pass) {
        const int jbase = blockIdx.y * 256 + pass * 16;
        __syncthreads();   // protect previous pass's xb_s readers
        for (int idx = t; idx < 16 * MLPH; idx += 256) {
            const int r = idx / MLPH, c = idx - r * MLPH;
            const int j = jbase + r;
            float v = 0.f;
            if (j >= 1 && j <= NN) v = XB[(size_t)(j - 1) * NPAD + c];
            xb_s[idx] = v;
        }
        __syncthreads();

        // Prefetch next pass's xb rows into cache (global_prefetch_b8).
        if (pass < 15 && iloc == 0) {
            const int jn = jbase + 16 + jslot;
            if (jn >= 1 && jn <= NN)
                __builtin_prefetch(XB + (size_t)(jn - 1) * NPAD, 0, 0);
        }

        const int j = jbase + jslot;
        if (j < JTOT) {
            const float* xbrow = xb_s + jslot * MLPH;
            float s = 0.f;
            #pragma unroll 5
            for (int k = 0; k < MLPH; k += 2) {
                const float2 xa2 = *(const float2*)(xarow + k);
                const float2 xb2 = *(const float2*)(xbrow + k);
                const float2 w2  = *(const float2*)(w2h_s + k);
                s += w2.x * tanh_fast(xa2.x + xb2.x);
                s += w2.y * tanh_fast(xa2.y + xb2.y);
            }
            raw_out[(size_t)(ibase + iloc) * JTOT + j] = s + base;
        }
    }
}

// ---------------------------------------------------------------------------
// Kernel C: per-row softmax (no max-shift: |raw| <= ~10.2, exp is safe) +
// per-row L1 loss partial vs one-hot.  Deterministic tree reduction.
// grid = 1024 rows, block = 256.
// ---------------------------------------------------------------------------
__global__ __launch_bounds__(256)
void softmax_loss_row(float* __restrict__ out,        // d_out (out[0]=loss)
                      const int* __restrict__ tgt,
                      float* __restrict__ lossRow)    // [1024]
{
    const int i = blockIdx.x;
    float* row = out + 1 + (size_t)i * JTOT;
    const int t = threadIdx.x;
    const int target = tgt[i];

    float ev[5];
    float lsum = 0.f, labs = 0.f;
    #pragma unroll
    for (int s = 0; s < 5; ++s) {
        const int j = t + s * 256;
        float e = 0.f;
        if (j < JTOT) {
            const float r = row[j];
            e = __expf(r);
            const float oh = (j == target) ? 1.f : 0.f;
            labs += fabsf(r - oh);
        }
        ev[s] = e;
        lsum += e;
    }

    __shared__ float red[256], red2[256];
    red[t] = lsum; red2[t] = labs;
    __syncthreads();
    for (int off = 128; off > 0; off >>= 1) {
        if (t < off) { red[t] += red[t + off]; red2[t] += red2[t + off]; }
        __syncthreads();
    }

    const float inv = 1.f / red[0];
    #pragma unroll
    for (int s = 0; s < 5; ++s) {
        const int j = t + s * 256;
        if (j < JTOT) row[j] = ev[s] * inv;
    }
    if (t == 0) lossRow[i] = red2[0];
}

// ---------------------------------------------------------------------------
// Kernel D: final deterministic loss reduction.  1 block, 256 threads.
// ---------------------------------------------------------------------------
__global__ __launch_bounds__(256)
void loss_final(const float* __restrict__ lossRow, float* __restrict__ out)
{
    const int t = threadIdx.x;
    float s = lossRow[t] + lossRow[t + 256] + lossRow[t + 512] + lossRow[t + 768];
    __shared__ float red[256];
    red[t] = s;
    __syncthreads();
    for (int off = 128; off > 0; off >>= 1) {
        if (t < off) red[t] += red[t + off];
        __syncthreads();
    }
    if (t == 0) out[0] = red[0] * (1.0f / ((float)NN * (float)JTOT));
}

// ---------------------------------------------------------------------------
extern "C" void kernel_launch(void* const* d_in, const int* in_sizes, int n_in,
                              void* d_out, int out_size, void* d_ws, size_t ws_size,
                              hipStream_t stream)
{
    const float* sentence = (const float*)d_in[0];   // [1024][1024]
    const int*   tgt      = (const int*)  d_in[1];   // [1024]
    const float* W1       = (const float*)d_in[2];   // [100][2048]
    const float* b1       = (const float*)d_in[3];   // [100]
    const float* W2       = (const float*)d_in[4];   // [100]
    const float* b2       = (const float*)d_in[5];   // [1]

    float* out = (float*)d_out;                      // [1 + 1024*1025]

    float* XA      = (float*)d_ws;                   // 1024*112
    float* XB      = XA + (size_t)NN * NPAD;         // 1024*112
    float* lossRow = XB + (size_t)NN * NPAD;         // 1024

    gemm_xab_wmma<<<dim3(64, 7, 2), 32, 0, stream>>>(sentence, W1, b1, XA, XB);
    raw_pass<<<dim3(64, 5), 256, 0, stream>>>(XA, XB, W2, b2, out + 1);
    softmax_loss_row<<<dim3(1024), 256, 0, stream>>>(out, tgt, lossRow);
    loss_final<<<1, 256, 0, stream>>>(lossRow, out);
}
